// AttentionHead_24919400251542
// MI455X (gfx1250) — matrix operand order repaired and checked by reading, not compiled, not run on hardware
//
#include <hip/hip_runtime.h>
#include <math.h>

// ---------------------------------------------------------------------------
// Problem constants (from the reference): B=4, T=4096, D=512, H=128.
// ---------------------------------------------------------------------------
#define kB_ 4
#define kT_ 4096
#define kD_ 512
#define kH_ 128
#define kM_ (kB_ * kT_)              // 16384 flattened rows of x
#define NEG_BIG (-1.0e30f)

typedef __attribute__((ext_vector_type(16))) _Float16 v16h;
typedef __attribute__((ext_vector_type(8)))  _Float16 v8h;
typedef __attribute__((ext_vector_type(8)))  float    v8f;
typedef __attribute__((ext_vector_type(4)))  int      v4i;

// gfx1250 async global->LDS copy path (guarded; falls back to sync b128 copy).
#if defined(__has_builtin)
#  if __has_builtin(__builtin_amdgcn_global_load_async_to_lds_b128)
#    define USE_ASYNC_COPY 1
#  endif
#endif
#ifndef USE_ASYNC_COPY
#  define USE_ASYNC_COPY 0
#endif

static __device__ __forceinline__ v8f wmma_f32_16x16x32(v16h a, v16h b, v8f c) {
  // (neg_a, A, neg_b, B, c_mod, C, reuse_a, reuse_b)
  return __builtin_amdgcn_wmma_f32_16x16x32_f16(false, a, false, b, (short)0, c,
                                                false, false);
}

// A fragment (16x32, M x K) from a row-major f32 matrix, converted to f16.
static __device__ __forceinline__ v16h load_a_from_f32(const float* src, int ld,
                                                       int row0, int k0, int lane) {
  const int lh = lane >> 4;
  const int lr = lane & 15;
  const float* p = src + (size_t)(row0 + lr) * ld + k0 + lh * 8;
  v8f lo = *(const v8f*)p;           // K = base .. base+7   (32B aligned)
  v8f hi = *(const v8f*)(p + 16);    // K = base+16 .. base+23
  v16h a;
#pragma unroll
  for (int i = 0; i < 8; ++i) { a[i] = (_Float16)lo[i]; a[8 + i] = (_Float16)hi[i]; }
  return a;
}

// A fragment from a row-major f16 matrix (global q tile or LDS P tile).
static __device__ __forceinline__ v16h load_a_from_f16(const _Float16* src, int ld,
                                                       int row0, int k0, int lane) {
  const int lh = lane >> 4;
  const int lr = lane & 15;
  const _Float16* p = src + (size_t)(row0 + lr) * ld + k0 + lh * 8;
  v8h lo = *(const v8h*)p;           // 16B-aligned -> b128
  v8h hi = *(const v8h*)(p + 16);
  return __builtin_shufflevector(lo, hi, 0, 1, 2, 3, 4, 5, 6, 7,
                                 8, 9, 10, 11, 12, 13, 14, 15);
}

// B fragment (32x16, K x N); element (k,n) = src[(n0+n)*ld + k0+k]
// ("n-major": contiguous in k -> two b128 loads). Lane half h holds
// K=8h..8h+7 and K=16+8h..16+8h+7 (same order as A), N = lane%16.
static __device__ __forceinline__ v16h load_b_nmajor_f16(const _Float16* src, int ld,
                                                         int k0, int n0, int lane) {
  const int lh = lane >> 4;
  const int lr = lane & 15;
  const _Float16* p = src + (size_t)(n0 + lr) * ld + k0 + lh * 8;
  v8h lo = *(const v8h*)p;
  v8h hi = *(const v8h*)(p + 16);
  return __builtin_shufflevector(lo, hi, 0, 1, 2, 3, 4, 5, 6, 7,
                                 8, 9, 10, 11, 12, 13, 14, 15);
}

// ---------------------------------------------------------------------------
// Stage one 64 x 128 f16 tile (16 KB) into LDS with 128 threads.
// Coalesced mapping: step i, thread t -> bytes [i*2048 + t*16, +16). Each async
// instruction's 32 lanes form one contiguous 512 B burst per wave.
// Async path: 8 global_load_async_to_lds_b128 per thread (ASYNCcnt-tracked).
// ---------------------------------------------------------------------------
static __device__ __forceinline__ void stage_tile(_Float16* dst, const _Float16* src,
                                                  int tid) {
#if USE_ASYNC_COPY
  typedef __attribute__((address_space(1))) v4i gv4i_t;   // global int4*
  typedef __attribute__((address_space(3))) v4i lv4i_t;   // LDS int4*
  gv4i_t* g = (gv4i_t*)src;
  lv4i_t* l = (lv4i_t*)dst;
#pragma unroll
  for (int i = 0; i < 8; ++i)
    __builtin_amdgcn_global_load_async_to_lds_b128(g + i * 128 + tid,
                                                   l + i * 128 + tid, 0, 0);
#else
  const v8h* s = (const v8h*)src;
  v8h* d = (v8h*)dst;
#pragma unroll
  for (int i = 0; i < 8; ++i) d[i * 128 + tid] = s[i * 128 + tid];
#endif
}

static __device__ __forceinline__ void wait_async_le8() {
#if USE_ASYNC_COPY
#  if __has_builtin(__builtin_amdgcn_s_wait_asynccnt)
  __builtin_amdgcn_s_wait_asynccnt(8);
#  else
  asm volatile("s_wait_asynccnt 8" ::: "memory");
#  endif
#endif
}

static __device__ __forceinline__ void wait_async_0() {
#if USE_ASYNC_COPY
#  if __has_builtin(__builtin_amdgcn_s_wait_asynccnt)
  __builtin_amdgcn_s_wait_asynccnt(0);
#  else
  asm volatile("s_wait_asynccnt 0" ::: "memory");
#  endif
#endif
}

// ---------------------------------------------------------------------------
// Kernel 0: one-time weight transpose+convert: w[D,H] f32 -> wt[H,D] f16.
// ---------------------------------------------------------------------------
__global__ __launch_bounds__(256) void transpose_w_kernel(
    const float* __restrict__ w, _Float16* __restrict__ wt) {
  const int idx = blockIdx.x * 256 + threadIdx.x;   // kD_*kH_ = 65536 threads
  const int k = idx >> 7;                           // / kH_
  const int h = idx & (kH_ - 1);
  wt[(size_t)h * kD_ + k] = (_Float16)w[idx];
}

// ---------------------------------------------------------------------------
// Kernel 1: outh = ((x @ w) + bias) * scale, stored f16.
// M = 16384, K = 512, N = 128. Block = 128 threads = 4 waves; wave owns a
// 16-row strip and all 8 N-subtiles. B-fragments come from the pre-transposed
// f16 weights (contiguous b128 loads, no in-loop conversion).
// transposed==0: outh[row, n] (row-major, for q/k).
// transposed==1: outh[b][n][t] (for v -> vt; one b128 store per subtile).
// ---------------------------------------------------------------------------
__global__ __launch_bounds__(128) void qkv_proj_kernel(
    const float* __restrict__ x, const _Float16* __restrict__ wt,
    const float* __restrict__ bias, _Float16* __restrict__ outh, float scale,
    int transposed) {
  const int lane = threadIdx.x & 31;
  const int wave = threadIdx.x >> 5;
  const int row0 = blockIdx.x * 64 + wave * 16;

  const v8f vzero = {0.f, 0.f, 0.f, 0.f, 0.f, 0.f, 0.f, 0.f};
  v8f acc[8];
#pragma unroll
  for (int i = 0; i < 8; ++i) acc[i] = vzero;

  for (int kc = 0; kc < kD_; kc += 32) {
    v16h a = load_a_from_f32(x, kD_, row0, kc, lane);
#pragma unroll
    for (int ns = 0; ns < 8; ++ns) {
      v16h b = load_b_nmajor_f16(wt, kD_, kc, ns * 16, lane);
      acc[ns] = wmma_f32_16x16x32(a, b, acc[ns]);
    }
  }

  const int lh = lane >> 4, lr = lane & 15;
  if (!transposed) {
#pragma unroll
    for (int ns = 0; ns < 8; ++ns) {
      const float bb = bias[ns * 16 + lr];
#pragma unroll
      for (int j = 0; j < 8; ++j) {
        const int row = row0 + lh * 8 + j;    // C layout: VGPR j -> M=j (+8 hi half)
        outh[(size_t)row * kH_ + ns * 16 + lr] = (_Float16)((acc[ns][j] + bb) * scale);
      }
    }
  } else {
    const int bidx = row0 / kT_;              // batch (tiles never straddle batches)
    const int t0 = row0 - bidx * kT_ + lh * 8;
    _Float16* ob = outh + (size_t)bidx * kH_ * kT_;
#pragma unroll
    for (int ns = 0; ns < 8; ++ns) {
      const float bb = bias[ns * 16 + lr];
      v8h pk;
#pragma unroll
      for (int j = 0; j < 8; ++j) pk[j] = (_Float16)((acc[ns][j] + bb) * scale);
      *(v8h*)(ob + (size_t)(ns * 16 + lr) * kT_ + t0) = pk;   // vt[h][t], b128
    }
  }
}

// ---------------------------------------------------------------------------
// Kernel 2: causal flash attention. q/k row-major f16 (q pre-scaled), v as
// transposed vt[b][h][t] f16. Block = 128 threads = 4 waves, 64 query rows;
// K tiles (64 x 128) double-buffered in LDS via async global->LDS copies.
// ---------------------------------------------------------------------------
__global__ __launch_bounds__(128) void flash_attn_kernel(
    const _Float16* __restrict__ qh, const _Float16* __restrict__ kh,
    const _Float16* __restrict__ vt, float* __restrict__ out) {
  __shared__ __align__(16) _Float16 kbuf[2][64 * kH_];   // 2 x 16 KB K tiles
  __shared__ __align__(16) _Float16 plds[4][16 * 64];    // per-wave P tiles (8 KB)

  const int tid = threadIdx.x;
  const int lane = tid & 31;
  const int wave = tid >> 5;
  const int lh = lane >> 4, lr = lane & 15;
  const int qbase = blockIdx.x * 64;
  const size_t bo = (size_t)blockIdx.y * kT_ * kH_;
  const _Float16* q = qh + bo;
  const _Float16* kmat = kh + bo;
  const _Float16* vtb = vt + bo;               // [H][T] for this batch
  const int row0 = qbase + wave * 16;

  // Q fragments for this wave's 16 rows (H=128 -> 4 K-chunks), reused all tiles.
  v16h aq[4];
#pragma unroll
  for (int hc = 0; hc < 4; ++hc) aq[hc] = load_a_from_f16(q, kH_, row0, hc * 32, lane);

  const v8f vzero = {0.f, 0.f, 0.f, 0.f, 0.f, 0.f, 0.f, 0.f};
  v8f oacc[8];                                 // 16 x 128 fp32 output accumulator
#pragma unroll
  for (int i = 0; i < 8; ++i) oacc[i] = vzero;
  float rm[8], rl[8];                          // per-row running max / sum
#pragma unroll
  for (int j = 0; j < 8; ++j) { rm[j] = NEG_BIG; rl[j] = 0.0f; }

  const int nTiles = qbase / 64 + 1;           // keys 0 .. qbase+63 (causal)
  stage_tile(&kbuf[0][0], kmat, tid);          // prologue: tile 0 in flight

  for (int kt = 0; kt < nTiles; ++kt) {
    const int kv0 = kt * 64;
    const _Float16* ldsK = &kbuf[kt & 1][0];
    const bool pf = (kt + 1 < nTiles);
    if (pf) {
      stage_tile(&kbuf[(kt + 1) & 1][0], kmat + (size_t)(kv0 + 64) * kH_, tid);
      __builtin_prefetch(vtb + (size_t)lane * 4 * kT_ + kv0 + 64, 0, 1);
      wait_async_le8();                        // tile kt done (async in-order)
    } else {
      wait_async_0();
    }
    __syncthreads();                           // K tile visible to all waves

    // ---- S = Q * K^T : 16 x 64, fp32 (B-fragments from LDS, ds_load_b128) ----
    v8f s[4];
#pragma unroll
    for (int ns = 0; ns < 4; ++ns) s[ns] = vzero;
#pragma unroll
    for (int hc = 0; hc < 4; ++hc) {
#pragma unroll
      for (int ns = 0; ns < 4; ++ns) {
        v16h bk = load_b_nmajor_f16(ldsK, kH_, hc * 32, ns * 16, lane);
        s[ns] = wmma_f32_16x16x32(aq[hc], bk, s[ns]);
      }
    }

    // ---- causal mask (only the diagonal tile can violate causality) ----
    if (kt == nTiles - 1) {
#pragma unroll
      for (int ns = 0; ns < 4; ++ns)
#pragma unroll
        for (int j = 0; j < 8; ++j) {
          const int qg = row0 + lh * 8 + j;
          const int kg = kv0 + ns * 16 + lr;
          s[ns][j] = (kg > qg) ? NEG_BIG : s[ns][j];
        }
    }

    // ---- online softmax: row max across 4 subtiles + 16 lanes of this half ----
    float alpha[8];
#pragma unroll
    for (int j = 0; j < 8; ++j) {
      float m = rm[j];
#pragma unroll
      for (int ns = 0; ns < 4; ++ns) m = fmaxf(m, s[ns][j]);
      m = fmaxf(m, __shfl_xor(m, 1, 16));
      m = fmaxf(m, __shfl_xor(m, 2, 16));
      m = fmaxf(m, __shfl_xor(m, 4, 16));
      m = fmaxf(m, __shfl_xor(m, 8, 16));
      alpha[j] = __expf(rm[j] - m);
      rm[j] = m;
    }
#pragma unroll
    for (int i = 0; i < 8; ++i)
#pragma unroll
      for (int j = 0; j < 8; ++j) oacc[i][j] *= alpha[j];

    // ---- P = exp(S - m): row sums + f16 P tile staged in LDS ----
    float psum[8];
#pragma unroll
    for (int j = 0; j < 8; ++j) psum[j] = 0.0f;
#pragma unroll
    for (int ns = 0; ns < 4; ++ns)
#pragma unroll
      for (int j = 0; j < 8; ++j) {
        const float p = __expf(s[ns][j] - rm[j]);
        psum[j] += p;
        plds[wave][(size_t)(lh * 8 + j) * 64 + ns * 16 + lr] = (_Float16)p;
      }
#pragma unroll
    for (int j = 0; j < 8; ++j) {
      float t = psum[j];
      t += __shfl_xor(t, 1, 16);
      t += __shfl_xor(t, 2, 16);
      t += __shfl_xor(t, 4, 16);
      t += __shfl_xor(t, 8, 16);
      rl[j] = rl[j] * alpha[j] + t;
    }

    // P tile is wave-private; LDS in-order per wave, make write->read explicit.
    asm volatile("s_wait_dscnt 0" ::: "memory");

    // ---- O += P * V : contraction over 64 keys; V from transposed vt (b128) ----
#pragma unroll
    for (int kc = 0; kc < 2; ++kc) {
      v16h ap = load_a_from_f16(&plds[wave][0], 64, 0, kc * 32, lane);
#pragma unroll
      for (int hs = 0; hs < 8; ++hs) {
        v16h bv = load_b_nmajor_f16(vtb, kT_, kv0 + kc * 32, hs * 16, lane);
        oacc[hs] = wmma_f32_16x16x32(ap, bv, oacc[hs]);
      }
    }

    __syncthreads();   // all waves done with kbuf[kt&1] before it is re-staged
  }

  // ---- epilogue: O / l, fp32 store ----
  float* o = out + bo;
#pragma unroll
  for (int hs = 0; hs < 8; ++hs)
#pragma unroll
    for (int j = 0; j < 8; ++j) {
      const int row = row0 + lh * 8 + j;
      o[(size_t)row * kH_ + hs * 16 + lr] = oacc[hs][j] / rl[j];
    }
}

// ---------------------------------------------------------------------------
// Host launcher. Inputs (setup_inputs order): x, mask, wq, bq, wk, bk, wv, bv.
// Causal mask computed analytically; d_in[1] unused.
// Workspace: qh | kh | vt (4 MiB each f16) | wtq | wtk | wtv (128 KiB each).
// ---------------------------------------------------------------------------
extern "C" void kernel_launch(void* const* d_in, const int* in_sizes, int n_in,
                              void* d_out, int out_size, void* d_ws, size_t ws_size,
                              hipStream_t stream) {
  (void)in_sizes; (void)n_in; (void)out_size; (void)ws_size;
  const float* x  = (const float*)d_in[0];
  const float* wq = (const float*)d_in[2];
  const float* bq = (const float*)d_in[3];
  const float* wk = (const float*)d_in[4];
  const float* bk = (const float*)d_in[5];
  const float* wv = (const float*)d_in[6];
  const float* bv = (const float*)d_in[7];
  float* out = (float*)d_out;

  _Float16* qh  = (_Float16*)d_ws;
  _Float16* kh  = qh + (size_t)kM_ * kH_;
  _Float16* vt  = kh + (size_t)kM_ * kH_;
  _Float16* wtq = vt + (size_t)kM_ * kH_;
  _Float16* wtk = wtq + (size_t)kD_ * kH_;
  _Float16* wtv = wtk + (size_t)kD_ * kH_;

  const float qscale = 0.08838834764831845f;   // 1/sqrt(H); folded into q

  const int wN = (kD_ * kH_) / 256;            // 256 blocks
  transpose_w_kernel<<<dim3(wN), dim3(256), 0, stream>>>(wq, wtq);
  transpose_w_kernel<<<dim3(wN), dim3(256), 0, stream>>>(wk, wtk);
  transpose_w_kernel<<<dim3(wN), dim3(256), 0, stream>>>(wv, wtv);

  qkv_proj_kernel<<<dim3(kM_ / 64), dim3(128), 0, stream>>>(x, wtq, bq, qh, qscale, 0);
  qkv_proj_kernel<<<dim3(kM_ / 64), dim3(128), 0, stream>>>(x, wtk, bk, kh, 1.0f, 0);
  qkv_proj_kernel<<<dim3(kM_ / 64), dim3(128), 0, stream>>>(x, wtv, bv, vt, 1.0f, 1);

  flash_attn_kernel<<<dim3(kT_ / 64, kB_), dim3(128), 0, stream>>>(qh, kh, vt, out);
}
